// NeuralGCDE_63660005261503
// MI455X (gfx1250) — matrix operand order repaired and checked by reading, not compiled, not run on hardware
//
#include <hip/hip_runtime.h>
#include <math.h>

typedef _Float16 half16 __attribute__((ext_vector_type(16)));
typedef float    float8 __attribute__((ext_vector_type(8)));
typedef unsigned int uint4v __attribute__((ext_vector_type(4)));
typedef int int4g __attribute__((vector_size(16)));   // matches builtin pointee

#define B_   32
#define N_   256
#define T_   13
#define CIN_ 2
#define HID_ 64
#define EMB_ 10
#define BN_  (B_ * N_)        /* 8192  */
#define SZ_  (BN_ * HID_)     /* 524288 */

// ---------------------------------------------------------------------------
// Async global->LDS staging (CDNA5). Guarded: falls back to b128 copies.
// ---------------------------------------------------------------------------
#if defined(__HIP_DEVICE_COMPILE__) && defined(__gfx1250__)
#  if __has_builtin(__builtin_amdgcn_global_load_async_to_lds_b128) && \
      __has_builtin(__builtin_amdgcn_s_wait_asynccnt)
#    define USE_ASYNC_LDS 1
#  endif
#endif
#ifndef USE_ASYNC_LDS
#  define USE_ASYNC_LDS 0
#endif

__device__ __forceinline__ void copy16B(_Float16* dst_lds, const _Float16* src_glb) {
#if USE_ASYNC_LDS
  __builtin_amdgcn_global_load_async_to_lds_b128(
      (__attribute__((address_space(1))) int4g*)(unsigned long long)(size_t)src_glb,
      (__attribute__((address_space(3))) int4g*)(unsigned int)(size_t)dst_lds,
      0, 0);
#else
  *(uint4v*)dst_lds = *(const uint4v*)src_glb;
#endif
}

__device__ __forceinline__ void staging_fence() {
#if USE_ASYNC_LDS
  __builtin_amdgcn_s_wait_asynccnt(0);
#endif
  __syncthreads();
}

// Copy `rows` rows of `chunks`*8 halves from global (stride sls halves) into
// LDS (stride dls halves). All bases/strides 16B-aligned.
__device__ __forceinline__ void copy_rows(_Float16* dst, int dls,
                                          const _Float16* src, size_t sls,
                                          int rows, int chunks, int tid) {
  for (int i = tid; i < rows * chunks; i += 256) {
    const int r = i / chunks, c = i % chunks;
    copy16B(dst + (size_t)r * dls + c * 8, src + (size_t)r * sls + c * 8);
  }
}

// ---------------------------------------------------------------------------
// WMMA helpers (CDNA5 16x16x32 f16 -> f32), fragment layouts per ISA 7.12.2
// ---------------------------------------------------------------------------
__device__ __forceinline__ float8 wmma16(half16 a, half16 b, float8 c) {
  return __builtin_amdgcn_wmma_f32_16x16x32_f16(false, a, false, b,
                                                (short)0, c, false, false);
}

__device__ __forceinline__ half16 frag_a(const _Float16* src, int ld, int lane) {
  const int m  = lane & 15;
  const int hf = lane >> 4;
  const _Float16* p0 = src + m * ld + (hf << 3);
  const _Float16* p1 = p0 + 16;
  half16 a;
#pragma unroll
  for (int e = 0; e < 8; ++e) { a[e] = p0[e]; a[e + 8] = p1[e]; }
  return a;
}

__device__ __forceinline__ half16 frag_bt(const _Float16* srcT, int ld, int lane) {
  const int nn = lane & 15;
  const int hf = lane >> 4;
  const _Float16* p = srcT + nn * ld + (hf << 3);
  half16 b;
#pragma unroll
  for (int e = 0; e < 8; ++e) { b[e] = p[e]; b[e + 8] = p[e + 16]; }
  return b;
}

#define ZERO8 {0.f,0.f,0.f,0.f,0.f,0.f,0.f,0.f}

// ---------------------------------------------------------------------------
// func_f fused 3-layer MLP. Weights pre-transposed f16: W1t/W2t [64][64],
// W3t [128][64]. X f32 (M,64) -> F f32 (M,128). 64 rows/WG, 8 waves.
// ---------------------------------------------------------------------------
__global__ __launch_bounds__(256) void funcf_kernel(
    const float* __restrict__ X,
    const _Float16* __restrict__ W1t, const float* __restrict__ b1,
    const _Float16* __restrict__ W2t, const float* __restrict__ b2,
    const _Float16* __restrict__ W3t, const float* __restrict__ b3,
    float* __restrict__ F)
{
  __shared__ __align__(16) _Float16 sAct0[64][64 + 8];
  __shared__ __align__(16) _Float16 sAct1[64][64 + 8];
  __shared__ __align__(16) _Float16 sW1t[64][64 + 8];
  __shared__ __align__(16) _Float16 sW2t[64][64 + 8];
  __shared__ __align__(16) _Float16 sW3t[128][64 + 8];
  __shared__ float sB1[64], sB2[64], sB3[128];

  const int row0 = blockIdx.x * 64;
  const int tid  = threadIdx.x;

  for (int i = tid; i < 64 * 64; i += 256) {
    const int r = i >> 6, c = i & 63;
    sAct0[r][c] = (_Float16)X[(size_t)(row0 + r) * 64 + c];
  }
  copy_rows(&sW1t[0][0], 72, W1t, 64, 64, 8, tid);
  copy_rows(&sW2t[0][0], 72, W2t, 64, 64, 8, tid);
  copy_rows(&sW3t[0][0], 72, W3t, 64, 128, 8, tid);
  if (tid < 64)  { sB1[tid] = b1[tid]; sB2[tid] = b2[tid]; }
  if (tid < 128) sB3[tid] = b3[tid];
  staging_fence();

  const int wave = tid >> 5, lane = tid & 31;
  const int nlane = lane & 15, hf = lane >> 4;

  // layer 1 -> relu -> sAct1
  for (int tt = 0; tt < 2; ++tt) {
    const int tile = wave * 2 + tt;
    const int m0 = (tile >> 2) * 16, n0 = (tile & 3) * 16;
    float8 acc = ZERO8;
    for (int k0 = 0; k0 < 64; k0 += 32)
      acc = wmma16(frag_a(&sAct0[m0][k0], 72, lane),
                   frag_bt(&sW1t[n0][k0], 72, lane), acc);
#pragma unroll
    for (int j = 0; j < 8; ++j) {
      float v = acc[j] + sB1[n0 + nlane];
      sAct1[m0 + (hf << 3) + j][n0 + nlane] = (_Float16)(v > 0.f ? v : 0.f);
    }
  }
  __syncthreads();

  // layer 2 -> relu -> sAct0
  for (int tt = 0; tt < 2; ++tt) {
    const int tile = wave * 2 + tt;
    const int m0 = (tile >> 2) * 16, n0 = (tile & 3) * 16;
    float8 acc = ZERO8;
    for (int k0 = 0; k0 < 64; k0 += 32)
      acc = wmma16(frag_a(&sAct1[m0][k0], 72, lane),
                   frag_bt(&sW2t[n0][k0], 72, lane), acc);
#pragma unroll
    for (int j = 0; j < 8; ++j) {
      float v = acc[j] + sB2[n0 + nlane];
      sAct0[m0 + (hf << 3) + j][n0 + nlane] = (_Float16)(v > 0.f ? v : 0.f);
    }
  }
  __syncthreads();

  // layer 3: 64x128 -> tanh -> global
  for (int tt = 0; tt < 4; ++tt) {
    const int tile = wave * 4 + tt;
    const int m0 = (tile >> 3) * 16, n0 = (tile & 7) * 16;
    float8 acc = ZERO8;
    for (int k0 = 0; k0 < 64; k0 += 32)
      acc = wmma16(frag_a(&sAct0[m0][k0], 72, lane),
                   frag_bt(&sW3t[n0][k0], 72, lane), acc);
#pragma unroll
    for (int j = 0; j < 8; ++j) {
      const int r = row0 + m0 + (hf << 3) + j;
      F[(size_t)r * 128 + n0 + nlane] = tanhf(acc[j] + sB3[n0 + nlane]);
    }
  }
}

// ---------------------------------------------------------------------------
// func_g projection: Xg = relu(Z@Wg_in+b). Emits f16 in two layouts:
//   Xg16  [bn][c]          (row-major, pernode A operand)
//   XgT16 [b][c][node]     (feature-major, graphconv B operand)
// ---------------------------------------------------------------------------
__global__ __launch_bounds__(256) void linear64_relu_kernel(
    const float* __restrict__ X, const _Float16* __restrict__ Wt,
    const float* __restrict__ bias,
    _Float16* __restrict__ Xg16, _Float16* __restrict__ XgT16)
{
  __shared__ __align__(16) _Float16 sX[64][64 + 8];
  __shared__ __align__(16) _Float16 sWt[64][64 + 8];
  __shared__ float sB[64];

  const int row0 = blockIdx.x * 64;
  const int tid  = threadIdx.x;
  for (int i = tid; i < 64 * 64; i += 256) {
    const int r = i >> 6, c = i & 63;
    sX[r][c] = (_Float16)X[(size_t)(row0 + r) * 64 + c];
  }
  copy_rows(&sWt[0][0], 72, Wt, 64, 64, 8, tid);
  if (tid < 64) sB[tid] = bias[tid];
  staging_fence();

  const int wave = tid >> 5, lane = tid & 31;
  const int nlane = lane & 15, hf = lane >> 4;

  for (int tt = 0; tt < 2; ++tt) {
    const int tile = wave * 2 + tt;
    const int m0 = (tile >> 2) * 16, n0 = (tile & 3) * 16;
    float8 acc = ZERO8;
    for (int k0 = 0; k0 < 64; k0 += 32)
      acc = wmma16(frag_a(&sX[m0][k0], 72, lane),
                   frag_bt(&sWt[n0][k0], 72, lane), acc);
#pragma unroll
    for (int j = 0; j < 8; ++j) {
      const int bn = row0 + m0 + (hf << 3) + j;
      const int c  = n0 + nlane;
      float v = acc[j] + sB[c];
      v = v > 0.f ? v : 0.f;
      const _Float16 hv = (_Float16)v;
      Xg16[(size_t)bn * 64 + c] = hv;
      const int b = bn >> 8, node = bn & 255;
      XgT16[(size_t)b * 16384 + (size_t)c * 256 + node] = hv;
    }
  }
}

// ---------------------------------------------------------------------------
// Graph conv: Yg16[b] = A(256x256) @ Xg[b](256x64); grid = B*4
// ---------------------------------------------------------------------------
__global__ __launch_bounds__(256) void graphconv_kernel(
    const _Float16* __restrict__ A16, const _Float16* __restrict__ XgT16,
    _Float16* __restrict__ Yg16)
{
  __shared__ __align__(16) _Float16 sA[64][64 + 8];
  __shared__ __align__(16) _Float16 sXt[64][64 + 8];

  const int b    = blockIdx.x >> 2;
  const int row0 = (blockIdx.x & 3) * 64;
  const int tid  = threadIdx.x;
  const int wave = tid >> 5, lane = tid & 31;
  const int nlane = lane & 15, hf = lane >> 4;

  float8 acc0 = ZERO8;
  float8 acc1 = ZERO8;
  const int tile0 = wave * 2;
  const int m0a = (tile0 >> 2) * 16,       n0a = (tile0 & 3) * 16;
  const int m0b = ((tile0 + 1) >> 2) * 16, n0b = ((tile0 + 1) & 3) * 16;

  for (int kc = 0; kc < 256; kc += 64) {
    __syncthreads();
    copy_rows(&sA[0][0], 72, A16 + (size_t)row0 * 256 + kc, 256, 64, 8, tid);
    copy_rows(&sXt[0][0], 72, XgT16 + (size_t)b * 16384 + kc, 256, 64, 8, tid);
    staging_fence();
    for (int k0 = 0; k0 < 64; k0 += 32) {
      acc0 = wmma16(frag_a(&sA[m0a][k0], 72, lane),
                    frag_bt(&sXt[n0a][k0], 72, lane), acc0);
      acc1 = wmma16(frag_a(&sA[m0b][k0], 72, lane),
                    frag_bt(&sXt[n0b][k0], 72, lane), acc1);
    }
  }
#pragma unroll
  for (int j = 0; j < 8; ++j) {
    Yg16[((size_t)b * 256 + row0 + m0a + (hf << 3) + j) * 64 + n0a + nlane] =
        (_Float16)acc0[j];
    Yg16[((size_t)b * 256 + row0 + m0b + (hf << 3) + j) * 64 + n0b + nlane] =
        (_Float16)acc1[j];
  }
}

// ---------------------------------------------------------------------------
// Per-node einsum + output projection (fused). One WG per node n.
//   mid = [Xg|Yg](32x128) @ WnT[n]^T(128x64) + bn[n]
//   G   = tanh(mid @ Wg_out + bg_out)   (32x128, f32 out)
// ---------------------------------------------------------------------------
__global__ __launch_bounds__(256) void pernode_kernel(
    const _Float16* __restrict__ Xg16, const _Float16* __restrict__ Yg16,
    const _Float16* __restrict__ WnT16, const float* __restrict__ bnv,
    const _Float16* __restrict__ WgOutT, const float* __restrict__ bg_out,
    float* __restrict__ G)
{
  __shared__ __align__(16) _Float16 sA[32][128 + 8];
  __shared__ __align__(16) _Float16 sWnt[64][128 + 8];
  __shared__ __align__(16) _Float16 sMid[32][64 + 8];
  __shared__ __align__(16) _Float16 sWgt[128][64 + 8];
  __shared__ float sBn[64], sBg[128];

  const int n   = blockIdx.x;
  const int tid = threadIdx.x;

  copy_rows(&sA[0][0],  136, Xg16 + (size_t)n * 64, 16384, 32, 8, tid);
  copy_rows(&sA[0][64], 136, Yg16 + (size_t)n * 64, 16384, 32, 8, tid);
  copy_rows(&sWnt[0][0], 136, WnT16 + (size_t)n * 8192, 128, 64, 16, tid);
  copy_rows(&sWgt[0][0], 72, WgOutT, 64, 128, 8, tid);
  if (tid < 64)  sBn[tid] = bnv[(size_t)n * 64 + tid];
  if (tid < 128) sBg[tid] = bg_out[tid];
  staging_fence();

  const int wave = tid >> 5, lane = tid & 31;
  const int nlane = lane & 15, hf = lane >> 4;

  // GEMM1: 32x128 @ 128x64 -> 8 tiles, one per wave
  {
    const int m0 = (wave >> 2) * 16, n0 = (wave & 3) * 16;
    float8 acc = ZERO8;
    for (int k0 = 0; k0 < 128; k0 += 32)
      acc = wmma16(frag_a(&sA[m0][k0], 136, lane),
                   frag_bt(&sWnt[n0][k0], 136, lane), acc);
#pragma unroll
    for (int j = 0; j < 8; ++j)
      sMid[m0 + (hf << 3) + j][n0 + nlane] = (_Float16)(acc[j] + sBn[n0 + nlane]);
  }
  __syncthreads();

  // GEMM2: 32x64 @ 64x128 -> 16 tiles, 2/wave; tanh -> global f32
  for (int tt = 0; tt < 2; ++tt) {
    const int tile = wave * 2 + tt;
    const int m0 = (tile >> 3) * 16, n0 = (tile & 7) * 16;
    float8 acc = ZERO8;
    for (int k0 = 0; k0 < 64; k0 += 32)
      acc = wmma16(frag_a(&sMid[m0][k0], 72, lane),
                   frag_bt(&sWgt[n0][k0], 72, lane), acc);
#pragma unroll
    for (int j = 0; j < 8; ++j) {
      const int bb = m0 + (hf << 3) + j;
      G[((size_t)bb * 256 + n) * 128 + n0 + nlane] = tanhf(acc[j] + sBg[n0 + nlane]);
    }
  }
}

// ---------------------------------------------------------------------------
// One-time setup kernels
// ---------------------------------------------------------------------------
__global__ void adj16_kernel(const float* __restrict__ Eg, _Float16* __restrict__ A16) {
  const int n = blockIdx.x * blockDim.x + threadIdx.x;
  if (n >= 256) return;
  float en[EMB_];
#pragma unroll
  for (int d = 0; d < EMB_; ++d) en[d] = Eg[n * EMB_ + d];
  float mx = 0.f;
  for (int m = 0; m < 256; ++m) {
    float s = 0.f;
#pragma unroll
    for (int d = 0; d < EMB_; ++d) s += en[d] * Eg[m * EMB_ + d];
    mx = fmaxf(mx, fmaxf(s, 0.f));
  }
  float sum = 0.f;
  for (int m = 0; m < 256; ++m) {
    float s = 0.f;
#pragma unroll
    for (int d = 0; d < EMB_; ++d) s += en[d] * Eg[m * EMB_ + d];
    sum += expf(fmaxf(s, 0.f) - mx);
  }
  for (int m = 0; m < 256; ++m) {
    float s = 0.f;
#pragma unroll
    for (int d = 0; d < EMB_; ++d) s += en[d] * Eg[m * EMB_ + d];
    A16[(size_t)n * 256 + m] = (_Float16)(expf(fmaxf(s, 0.f) - mx) / sum);
  }
}

// WnT16[n][o][ki] = sum_d Eg[n,d] * W_pool[d][ki][o]   (ki = k*64+i, 0..127)
__global__ void wnT16_kernel(const float* __restrict__ Eg,
                             const float* __restrict__ W_pool,
                             _Float16* __restrict__ WnT16) {
  const int gid = blockIdx.x * blockDim.x + threadIdx.x;
  if (gid >= 256 * 8192) return;
  const int n  = gid >> 13;
  const int o  = (gid >> 7) & 63;
  const int ki = gid & 127;
  float s = 0.f;
#pragma unroll
  for (int d = 0; d < EMB_; ++d)
    s += Eg[n * EMB_ + d] * W_pool[(size_t)d * 8192 + ki * 64 + o];
  WnT16[gid] = (_Float16)s;
}

__global__ void bnv_kernel(const float* __restrict__ Eg,
                           const float* __restrict__ b_pool,
                           float* __restrict__ bnv) {
  const int gid = blockIdx.x * blockDim.x + threadIdx.x;
  if (gid >= 256 * 64) return;
  const int n = gid >> 6, o = gid & 63;
  float s = 0.f;
#pragma unroll
  for (int d = 0; d < EMB_; ++d)
    s += Eg[n * EMB_ + d] * b_pool[d * 64 + o];
  bnv[gid] = s;
}

// dst[N][K] f16 = transpose of src[K][N] f32
__global__ void cvtT_kernel(const float* __restrict__ src,
                            _Float16* __restrict__ dst, int K, int N) {
  const int gid = blockIdx.x * blockDim.x + threadIdx.x;
  if (gid >= K * N) return;
  const int r = gid / N, c = gid % N;
  dst[(size_t)c * K + r] = (_Float16)src[gid];
}

__global__ void init_kernel(const float* __restrict__ coeff_a,
                            const float* __restrict__ W_h, const float* __restrict__ b_h,
                            const float* __restrict__ W_z, const float* __restrict__ b_z,
                            float* __restrict__ Hb, float* __restrict__ Zb,
                            float* __restrict__ out) {
  const int gid = blockIdx.x * blockDim.x + threadIdx.x;
  if (gid >= SZ_) return;
  const int bn = gid >> 6, h = gid & 63;
  const float x0 = coeff_a[(size_t)bn * ((T_ - 1) * CIN_) + 0];
  const float x1 = coeff_a[(size_t)bn * ((T_ - 1) * CIN_) + 1];
  const float hv = x0 * W_h[h] + x1 * W_h[64 + h] + b_h[h];
  const float zv = x0 * W_z[h] + x1 * W_z[64 + h] + b_z[h];
  Hb[gid] = hv;
  Zb[gid] = zv;
  out[((size_t)bn * T_ + 0) * 64 + h] = zv;
}

// ---------------------------------------------------------------------------
// RK stage combine (3/8 rule)
// ---------------------------------------------------------------------------
__global__ void update_kernel(
    const float* __restrict__ F, const float* __restrict__ G,
    const float* __restrict__ cb, const float* __restrict__ cc,
    const float* __restrict__ cd,
    int idx, float frac, float dt, int mode, int tout,
    float* __restrict__ Hb, float* __restrict__ Zb,
    float* __restrict__ Hs, float* __restrict__ Zs,
    float* __restrict__ K1h, float* __restrict__ K1z,
    float* __restrict__ K2h, float* __restrict__ K2z,
    float* __restrict__ K3h, float* __restrict__ K3z,
    float* __restrict__ out)
{
  const int gid = blockIdx.x * blockDim.x + threadIdx.x;
  if (gid >= SZ_) return;
  const int bn = gid >> 6, h = gid & 63;
  const size_t cbase = (size_t)bn * ((T_ - 1) * CIN_) + (size_t)idx * CIN_;
  const float dX0 = cb[cbase]     + (cc[cbase]     + cd[cbase]     * frac) * frac;
  const float dX1 = cb[cbase + 1] + (cc[cbase + 1] + cd[cbase + 1] * frac) * frac;
  const size_t fbase = (size_t)bn * 128 + (size_t)h * 2;
  const float f0 = F[fbase], f1 = F[fbase + 1];
  const float g0 = G[fbase], g1 = G[fbase + 1];
  const float dh = f0 * dX0 + f1 * dX1;
  const float dz = g0 * f0 * dX0 + g1 * f1 * dX1;
  const float third = 1.0f / 3.0f;

  if (mode == 1) {
    K1h[gid] = dh; K1z[gid] = dz;
    Hs[gid] = Hb[gid] + dt * third * dh;
    Zs[gid] = Zb[gid] + dt * third * dz;
  } else if (mode == 2) {
    K2h[gid] = dh; K2z[gid] = dz;
    Hs[gid] = Hb[gid] + dt * (dh - third * K1h[gid]);
    Zs[gid] = Zb[gid] + dt * (dz - third * K1z[gid]);
  } else if (mode == 3) {
    K3h[gid] = dh; K3z[gid] = dz;
    Hs[gid] = Hb[gid] + dt * (K1h[gid] - K2h[gid] + dh);
    Zs[gid] = Zb[gid] + dt * (K1z[gid] - K2z[gid] + dz);
  } else {
    const float hn = Hb[gid] + dt * 0.125f * (K1h[gid] + 3.f * (K2h[gid] + K3h[gid]) + dh);
    const float zn = Zb[gid] + dt * 0.125f * (K1z[gid] + 3.f * (K2z[gid] + K3z[gid]) + dz);
    Hb[gid] = hn; Zb[gid] = zn;
    out[((size_t)bn * T_ + tout) * 64 + h] = zn;
  }
}

// ---------------------------------------------------------------------------
// Host orchestration
// ---------------------------------------------------------------------------
extern "C" void kernel_launch(void* const* d_in, const int* in_sizes, int n_in,
                              void* d_out, int out_size, void* d_ws, size_t ws_size,
                              hipStream_t stream) {
  (void)in_sizes; (void)n_in; (void)out_size; (void)ws_size;

  const float* coeff_a = (const float*)d_in[1];
  const float* coeff_b = (const float*)d_in[2];
  const float* coeff_c = (const float*)d_in[3];
  const float* coeff_d = (const float*)d_in[4];
  const float* W_h    = (const float*)d_in[5];
  const float* b_h    = (const float*)d_in[6];
  const float* W_z    = (const float*)d_in[7];
  const float* b_z    = (const float*)d_in[8];
  const float* Wf_in  = (const float*)d_in[9];
  const float* bf_in  = (const float*)d_in[10];
  const float* Wf_hid = (const float*)d_in[11];
  const float* bf_hid = (const float*)d_in[12];
  const float* Wf_out = (const float*)d_in[13];
  const float* bf_out = (const float*)d_in[14];
  const float* Wg_in  = (const float*)d_in[15];
  const float* bg_in  = (const float*)d_in[16];
  const float* Eg     = (const float*)d_in[17];
  const float* W_pool = (const float*)d_in[18];
  const float* b_pool = (const float*)d_in[19];
  const float* Wg_out = (const float*)d_in[20];
  const float* bg_out = (const float*)d_in[21];
  float* out = (float*)d_out;

  float* ws = (float*)d_ws;
  float* Hb  = ws; ws += SZ_;
  float* Zb  = ws; ws += SZ_;
  float* Hs  = ws; ws += SZ_;
  float* Zs  = ws; ws += SZ_;
  float* K1h = ws; ws += SZ_;
  float* K1z = ws; ws += SZ_;
  float* K2h = ws; ws += SZ_;
  float* K2z = ws; ws += SZ_;
  float* K3h = ws; ws += SZ_;
  float* K3z = ws; ws += SZ_;
  float* Fb  = ws; ws += 2 * SZ_;
  float* Gb  = ws; ws += 2 * SZ_;
  float* bnv = ws; ws += 256 * 64;

  _Float16* hp = (_Float16*)ws;
  _Float16* Xg16    = hp; hp += SZ_;
  _Float16* XgT16   = hp; hp += SZ_;
  _Float16* Yg16    = hp; hp += SZ_;
  _Float16* Aadj16  = hp; hp += 256 * 256;
  _Float16* WnT16   = hp; hp += 256 * 128 * 64;
  _Float16* WfInT   = hp; hp += 64 * 64;
  _Float16* WfHidT  = hp; hp += 64 * 64;
  _Float16* WgInT   = hp; hp += 64 * 64;
  _Float16* WfOutT  = hp; hp += 128 * 64;
  _Float16* WgOutT  = hp; hp += 128 * 64;

  adj16_kernel<<<1, 256, 0, stream>>>(Eg, Aadj16);
  wnT16_kernel<<<(256 * 8192) / 256, 256, 0, stream>>>(Eg, W_pool, WnT16);
  bnv_kernel<<<(256 * 64) / 256, 256, 0, stream>>>(Eg, b_pool, bnv);
  cvtT_kernel<<<16, 256, 0, stream>>>(Wf_in, WfInT, 64, 64);
  cvtT_kernel<<<16, 256, 0, stream>>>(Wf_hid, WfHidT, 64, 64);
  cvtT_kernel<<<16, 256, 0, stream>>>(Wg_in, WgInT, 64, 64);
  cvtT_kernel<<<32, 256, 0, stream>>>(Wf_out, WfOutT, 64, 128);
  cvtT_kernel<<<32, 256, 0, stream>>>(Wg_out, WgOutT, 64, 128);
  init_kernel<<<SZ_ / 256, 256, 0, stream>>>(coeff_a, W_h, b_h, W_z, b_z, Hb, Zb, out);

  for (int s = 0; s < T_ - 1; ++s) {
    const float t0 = (float)s;
    const float dt = 1.0f;
    for (int stage = 1; stage <= 4; ++stage) {
      const float* Hin = (stage == 1) ? Hb : Hs;
      const float* Zin = (stage == 1) ? Zb : Zs;

      funcf_kernel<<<BN_ / 64, 256, 0, stream>>>(
          Hin, WfInT, bf_in, WfHidT, bf_hid, WfOutT, bf_out, Fb);

      linear64_relu_kernel<<<BN_ / 64, 256, 0, stream>>>(
          Zin, WgInT, bg_in, Xg16, XgT16);
      graphconv_kernel<<<B_ * 4, 256, 0, stream>>>(Aadj16, XgT16, Yg16);
      pernode_kernel<<<N_, 256, 0, stream>>>(
          Xg16, Yg16, WnT16, bnv, WgOutT, bg_out, Gb);

      float tt;
      if (stage == 1)      tt = t0;
      else if (stage == 2) tt = t0 + dt * (1.0f / 3.0f);
      else if (stage == 3) tt = t0 + dt * (2.0f / 3.0f);
      else                 tt = t0 + dt;
      int idx = (int)floorf(tt);
      if (idx > T_ - 2) idx = T_ - 2;
      if (idx < 0) idx = 0;
      const float frac = tt - (float)idx;

      update_kernel<<<SZ_ / 256, 256, 0, stream>>>(
          Fb, Gb, coeff_b, coeff_c, coeff_d, idx, frac, dt, stage, s + 1,
          Hb, Zb, Hs, Zs, K1h, K1z, K2h, K2z, K3h, K3z, out);
    }
  }
}
